// GCN_73701638799910
// MI455X (gfx1250) — matrix operand order repaired and checked by reading, not compiled, not run on hardware
//
#include <hip/hip_runtime.h>
#include <math.h>

// ---------------- problem constants (match reference) ----------------
#define BB 32
#define NO 36
#define NA 3
#define NR 128
#define DD 1024
#define AAx 512
#define HH 2
#define EPSV 1e-8f

typedef __attribute__((ext_vector_type(16))) __bf16 v16bf;
typedef __attribute__((ext_vector_type(8)))  __bf16 v8bf;
typedef __attribute__((ext_vector_type(8)))  float  v8f;

// float -> bf16 bits, round-to-nearest-even
__device__ __forceinline__ unsigned short f2bf_bits(float f) {
  union { float f; unsigned int u; } in; in.f = f;
  unsigned int r = in.u + 0x7FFFu + ((in.u >> 16) & 1u);
  return (unsigned short)(r >> 16);
}

// ---------------------------------------------------------------------
// t_sro = [gather(obj_e, edges[:,0]) | rela | gather(obj_e, edges[:,1])]
// bf16, row-major (B*Nr, 3D)
// ---------------------------------------------------------------------
__global__ void pack_tsro_kernel(const float* __restrict__ obj_e,
                                 const float* __restrict__ rela,
                                 const int*   __restrict__ edges,
                                 unsigned short* __restrict__ tsro, int total) {
  int idx = blockIdx.x * blockDim.x + threadIdx.x;
  if (idx >= total) return;
  int c = idx % (3 * DD);
  int r = (idx / (3 * DD)) % NR;
  int b = idx / (3 * DD * NR);
  float v;
  if (c < DD) {
    int s = edges[(b * NR + r) * 2 + 0];
    v = obj_e[((long)b * NO + s) * DD + c];
  } else if (c < 2 * DD) {
    v = rela[((long)b * NR + r) * DD + (c - DD)];
  } else {
    int o = edges[(b * NR + r) * 2 + 1];
    v = obj_e[((long)b * NO + o) * DD + (c - 2 * DD)];
  }
  tsro[idx] = f2bf_bits(v);
}

// x_attr = [attr | obj_i_exp], x_obji = [obj_i_exp | attr], bf16 (3456 x 2D)
__global__ void pack_attr_kernel(const float* __restrict__ attr,
                                 const float* __restrict__ obj_i,
                                 unsigned short* __restrict__ x_attr,
                                 unsigned short* __restrict__ x_obji, int total) {
  int idx = blockIdx.x * blockDim.x + threadIdx.x;
  if (idx >= total) return;
  int c   = idx % (2 * DD);
  int row = idx / (2 * DD);        // (b*No+n)*Na+k
  int nob = row / NA;              // b*No+n
  if (c < DD) {
    x_attr[idx] = f2bf_bits(attr[(long)row * DD + c]);
    x_obji[idx] = f2bf_bits(obj_i[(long)nob * DD + c]);
  } else {
    int cc = c - DD;
    x_attr[idx] = f2bf_bits(obj_i[(long)nob * DD + cc]);
    x_obji[idx] = f2bf_bits(attr[(long)row * DD + cc]);
  }
}

// Pack fp32 weight (Z x K x N row-major) into this kernel's B-fragment bf16 layout (contiguous k):
// out[z][t][n][i] with t = k/16, i = k%16 -> per-lane B fragment is
// 16 contiguous bf16 (32 bytes).
__global__ void pack_w_kernel(const float* __restrict__ W,
                              unsigned short* __restrict__ out,
                              int K, int N, long total) {
  long idx = (long)blockIdx.x * blockDim.x + threadIdx.x;
  if (idx >= total) return;
  long zkn = (long)K * N;
  int  z   = (int)(idx / zkn);
  long rem = idx % zkn;
  int  t = (int)(rem / ((long)N * 16));
  int  r2 = (int)(rem % ((long)N * 16));
  int  n = r2 >> 4;
  int  i = r2 & 15;
  int  k = t * 16 + i;
  out[idx] = f2bf_bits(W[(long)z * zkn + (long)k * N + n]);
}

// flat fp32 -> bf16
__global__ void cvt_bf16_kernel(const float* __restrict__ in,
                                unsigned short* __restrict__ out, long total) {
  long idx = (long)blockIdx.x * blockDim.x + threadIdx.x;
  if (idx >= total) return;
  out[idx] = f2bf_bits(in[idx]);
}

// ---------------------------------------------------------------------
// WMMA GEMM: C[z] = epi(A @ B[z] + bias[z]).
// A: bf16 row-major M x K. Bp: packed bf16 [K/16][N][16] per z.
// One wave computes a 16x32 C tile (two accumulators share one A frag).
// Layouts used here (contiguous k order, same for A and B):
//   A frag  : lane l16 = row; per lane 16 contiguous elements
//             K in [16h,16h+16)  (h = lane>>4)
//   B frag  : lane l16 = col; 16 contiguous packed elements K=16h..16h+15
//   C frag  : VGPR r -> row = 8h + r, col = l16
// Epilogue flags are compile-time: no branches in the hot path.
// ---------------------------------------------------------------------
template<int ACT, int HAS_RES, int HAS_SCALE>
__global__ void gemm_wmma_kernel(const unsigned short* __restrict__ Abf,
                                 const unsigned short* __restrict__ Bpk,
                                 const float* __restrict__ bias,
                                 const float* __restrict__ residual,
                                 const float* __restrict__ row_scale,
                                 float* __restrict__ C,
                                 int N, int K,
                                 int tiles_n2, int total_tiles,
                                 long bz, long biasz, long cz)
{
  int wave = threadIdx.x >> 5;
  int tile = blockIdx.x * 8 + wave;
  if (tile >= total_tiles) return;          // wave-uniform: EXEC all-ones for WMMA
  int z    = blockIdx.z;
  int tm   = tile / tiles_n2;
  int tn2  = tile % tiles_n2;
  int lane = threadIdx.x & 31;
  int half = lane >> 4;
  int l16  = lane & 15;
  int col  = tn2 * 32 + l16;

  const __bf16* arow  = reinterpret_cast<const __bf16*>(Abf) +
                        ((long)(tm * 16 + l16)) * K;
  const __bf16* bbase = reinterpret_cast<const __bf16*>(Bpk) + (long)z * bz;

  v8f c0 = {}, c1 = {};
#pragma unroll 2
  for (int k0 = 0; k0 < K; k0 += 32) {
    // A fragment: two contiguous 16B loads
    v8bf a0 = *reinterpret_cast<const v8bf*>(arow + k0 + 16 * half);
    v8bf a1 = *reinterpret_cast<const v8bf*>(arow + k0 + 16 * half + 8);
    v16bf afr = __builtin_shufflevector(a0, a1, 0, 1, 2, 3, 4, 5, 6, 7,
                                        8, 9, 10, 11, 12, 13, 14, 15);
    // B fragments: contiguous 32B per column in packed layout
    const __bf16* bcol = bbase + (((long)((k0 >> 4) + half) * N + col) << 4);
    v16bf b0 = *reinterpret_cast<const v16bf*>(bcol);
    v16bf b1 = *reinterpret_cast<const v16bf*>(bcol + (16 << 4));
    c0 = __builtin_amdgcn_wmma_f32_16x16x32_bf16(false, afr, false, b0,
                                                 (short)0, c0, false, false);
    c1 = __builtin_amdgcn_wmma_f32_16x16x32_bf16(false, afr, false, b1,
                                                 (short)0, c1, false, false);
  }

  float bv0 = bias[(long)z * biasz + col];
  float bv1 = bias[(long)z * biasz + col + 16];
#pragma unroll
  for (int r = 0; r < 8; ++r) {
    int row = tm * 16 + 8 * half + r;
    long off0 = (long)row * N + col;
    float v0 = c0[r] + bv0;
    float v1 = c1[r] + bv1;
    if (ACT) { v0 = fmaxf(v0, 0.0f); v1 = fmaxf(v1, 0.0f); }
    if (HAS_RES) { v0 += residual[off0]; v1 += residual[off0 + 16]; }
    if (HAS_SCALE) { float s = row_scale[row]; v0 *= s; v1 *= s; }
    C[(long)z * cz + off0]      = v0;
    C[(long)z * cz + off0 + 16] = v1;
  }
}

// ---------------------------------------------------------------------
// logit_e[b,h,n,r] = sum_a tanh(p_e[b,r,a] + q_e[h,bn,a]) * w_ra[h,a] + b_ra[h]
// Block per (b,h,n); q + w rows cached in LDS; 8 waves sweep r.
// ---------------------------------------------------------------------
__global__ void logit_e_kernel(const float* __restrict__ p_e,
                               const float* __restrict__ q_e,
                               const float* __restrict__ w_ra,
                               const float* __restrict__ b_ra,
                               float* __restrict__ logit) {
  __shared__ float qs[AAx];
  __shared__ float ws[AAx];
  int idx = blockIdx.x;
  int n = idx % NO;
  int h = (idx / NO) % HH;
  int b = idx / (NO * HH);
  const float* qrow = q_e + ((long)h * BB * NO + (long)b * NO + n) * AAx;
  for (int a = threadIdx.x; a < AAx; a += blockDim.x) {
    qs[a] = qrow[a];
    ws[a] = w_ra[h * AAx + a];
  }
  __syncthreads();
  int wave = threadIdx.x >> 5, lane = threadIdx.x & 31;
  float bh = b_ra[h];
  for (int r = wave; r < NR; r += 8) {
    const float* pe = p_e + ((long)b * NR + r) * AAx;
    float acc = 0.0f;
    for (int a = lane; a < AAx; a += 32) acc += tanhf(pe[a] + qs[a]) * ws[a];
#pragma unroll
    for (int off = 16; off > 0; off >>= 1) acc += __shfl_xor(acc, off, 32);
    if (lane == 0) logit[(((long)b * HH + h) * NO + n) * NR + r] = acc + bh;
  }
}

// logit_i[b,h,n,k] over Na=3; block (128 thr) per (b,h,n), waves 0..2 handle k.
__global__ void logit_i_kernel(const float* __restrict__ p_i,
                               const float* __restrict__ q_i,
                               const float* __restrict__ w_aa,
                               const float* __restrict__ b_aa,
                               float* __restrict__ logit) {
  __shared__ float qs[AAx];
  __shared__ float ws[AAx];
  int idx = blockIdx.x;
  int n = idx % NO;
  int h = (idx / NO) % HH;
  int b = idx / (NO * HH);
  const float* qrow = q_i + ((long)h * BB * NO + (long)b * NO + n) * AAx;
  for (int a = threadIdx.x; a < AAx; a += blockDim.x) {
    qs[a] = qrow[a];
    ws[a] = w_aa[h * AAx + a];
  }
  __syncthreads();
  int wave = threadIdx.x >> 5, lane = threadIdx.x & 31;
  if (wave < NA) {
    int k = wave;
    const float* pi = p_i + (((long)b * NO + n) * NA + k) * AAx;
    float acc = 0.0f;
    for (int a = lane; a < AAx; a += 32) acc += tanhf(pi[a] + qs[a]) * ws[a];
#pragma unroll
    for (int off = 16; off > 0; off >>= 1) acc += __shfl_xor(acc, off, 32);
    if (lane == 0) logit[(((long)b * HH + h) * NO + n) * NA + k] = acc + b_aa[h];
  }
}

// softmax over Na=3 (+mask, renormalize), in-place on wbuf.
__global__ void softmax_attr_kernel(const float* __restrict__ attr_masks,
                                    float* __restrict__ wbuf) {
  int idx = blockIdx.x * blockDim.x + threadIdx.x;
  if (idx >= BB * HH * NO) return;
  int n = idx % NO;
  int h = (idx / NO) % HH;
  int b = idx / (NO * HH);
  long base = (((long)b * HH + h) * NO + n) * NA;
  float l0 = wbuf[base], l1 = wbuf[base + 1], l2 = wbuf[base + 2];
  float m = fmaxf(l0, fmaxf(l1, l2));
  float e0 = expf(l0 - m), e1 = expf(l1 - m), e2 = expf(l2 - m);
  float s = e0 + e1 + e2;
  long mb = ((long)b * NO + n) * NA;
  float w0 = e0 / s * attr_masks[mb];
  float w1 = e1 / s * attr_masks[mb + 1];
  float w2 = e2 / s * attr_masks[mb + 2];
  float s2 = w0 + w1 + w2 + EPSV;
  wbuf[base] = w0 / s2; wbuf[base + 1] = w1 / s2; wbuf[base + 2] = w2 / s2;
}

// softmax over Nr=128 (+mask, renormalize), wave per (b,h,n), in-place.
__global__ void softmax_rela_kernel(const float* __restrict__ obj_rela_masks,
                                    float* __restrict__ wbuf) {
  int gwave = (blockIdx.x * blockDim.x + threadIdx.x) >> 5;
  int lane = threadIdx.x & 31;
  if (gwave >= BB * HH * NO) return;
  int n = gwave % NO;
  int h = (gwave / NO) % HH;
  int b = gwave / (NO * HH);
  long base = (((long)b * HH + h) * NO + n) * NR;
  float l[4];
  float m = -1e30f;
#pragma unroll
  for (int i = 0; i < 4; ++i) { l[i] = wbuf[base + lane + 32 * i]; m = fmaxf(m, l[i]); }
#pragma unroll
  for (int off = 16; off > 0; off >>= 1) m = fmaxf(m, __shfl_xor(m, off, 32));
  float s = 0.0f;
#pragma unroll
  for (int i = 0; i < 4; ++i) { l[i] = expf(l[i] - m); s += l[i]; }
#pragma unroll
  for (int off = 16; off > 0; off >>= 1) s += __shfl_xor(s, off, 32);
  long mb = ((long)b * NO + n) * NR;
  float s2 = 0.0f;
#pragma unroll
  for (int i = 0; i < 4; ++i) {
    l[i] = l[i] / s * obj_rela_masks[mb + lane + 32 * i];
    s2 += l[i];
  }
#pragma unroll
  for (int off = 16; off > 0; off >>= 1) s2 += __shfl_xor(s2, off, 32);
  s2 += EPSV;
#pragma unroll
  for (int i = 0; i < 4; ++i) wbuf[base + lane + 32 * i] = l[i] / s2;
}

// new_obj_i = mean_h sum_k w_i * t_i + obj_i   (head-mean folded: 0.5*(w0+w1))
__global__ void newobj_i_kernel(const float* __restrict__ w_i,
                                const float* __restrict__ t_i,
                                const float* __restrict__ obj_i,
                                float* __restrict__ out) {
  int idx = blockIdx.x * blockDim.x + threadIdx.x;
  if (idx >= BB * NO * DD) return;
  int d = idx % DD;
  int n = (idx / DD) % NO;
  int b = idx / (DD * NO);
  float acc = 0.0f;
#pragma unroll
  for (int k = 0; k < NA; ++k) {
    float ww = 0.5f * (w_i[(((long)b * HH + 0) * NO + n) * NA + k] +
                       w_i[(((long)b * HH + 1) * NO + n) * NA + k]);
    acc += ww * t_i[(((long)b * NO + n) * NA + k) * DD + d];
  }
  out[((long)b * NO + n) * DD + d] = acc + obj_i[((long)b * NO + n) * DD + d];
}

// new_obj_e = mean_h sum_r w_e * t_e + obj_e; block per (b,n), w row in LDS.
__global__ void newobj_e_kernel(const float* __restrict__ w_e,
                                const float* __restrict__ t_e,
                                const float* __restrict__ obj_e,
                                float* __restrict__ out) {
  __shared__ float ws[NR];
  int n = blockIdx.x % NO;
  int b = blockIdx.x / NO;
  if (threadIdx.x < NR) {
    int r = threadIdx.x;
    ws[r] = 0.5f * (w_e[(((long)b * HH + 0) * NO + n) * NR + r] +
                    w_e[(((long)b * HH + 1) * NO + n) * NR + r]);
  }
  __syncthreads();
  for (int d = threadIdx.x; d < DD; d += blockDim.x) {
    float acc = 0.0f;
    for (int r = 0; r < NR; ++r) acc += ws[r] * t_e[((long)b * NR + r) * DD + d];
    out[((long)b * NO + n) * DD + d] = acc + obj_e[((long)b * NO + n) * DD + d];
  }
}

// ---------------------------------------------------------------------
static inline int cdiv(long a, long b) { return (int)((a + b - 1) / b); }

extern "C" void kernel_launch(void* const* d_in, const int* in_sizes, int n_in,
                              void* d_out, int out_size, void* d_ws, size_t ws_size,
                              hipStream_t stream) {
  const float* attr_vecs      = (const float*)d_in[0];
  const float* obj_vecs_i     = (const float*)d_in[1];
  const float* obj_vecs_e     = (const float*)d_in[2];
  const float* rela_vecs      = (const float*)d_in[3];
  const float* rela_masks     = (const float*)d_in[4];
  const float* obj_rela_masks = (const float*)d_in[5];
  const float* attr_masks     = (const float*)d_in[6];
  const float* W_rela = (const float*)d_in[7];   const float* b_rela = (const float*)d_in[8];
  const float* W_attr = (const float*)d_in[9];   const float* b_attr = (const float*)d_in[10];
  const float* W_obji = (const float*)d_in[11];  const float* b_obji = (const float*)d_in[12];
  const float* W_obje = (const float*)d_in[13];  const float* b_obje = (const float*)d_in[14];
  const float* W_ctx_attr = (const float*)d_in[15]; const float* b_ctx_attr = (const float*)d_in[16];
  const float* W_ctx_rela = (const float*)d_in[17]; const float* b_ctx_rela = (const float*)d_in[18];
  const float* W_ah = (const float*)d_in[19];    const float* b_ah = (const float*)d_in[20];
  const float* w_aa = (const float*)d_in[21];    const float* b_aa = (const float*)d_in[22];
  const float* W_rh = (const float*)d_in[23];    const float* b_rh = (const float*)d_in[24];
  const float* w_ra = (const float*)d_in[25];    const float* b_ra = (const float*)d_in[26];
  const int*   edges = (const int*)d_in[27];

  // outputs, concatenated flat in return order
  float* out_attr = (float*)d_out;                         // (B, No*Na, D)
  float* out_obji = out_attr + (long)BB * NO * NA * DD;    // (B, No, D)
  float* out_obje = out_obji + (long)BB * NO * DD;         // (B, No, D)
  float* out_rela = out_obje + (long)BB * NO * DD;         // (B, Nr, D)

  // workspace carving
  char* wp = (char*)d_ws;
  auto carve = [&](size_t bytes) -> char* {
    char* p = wp;
    wp += (bytes + 255) & ~(size_t)255;
    return p;
  };
  const long M_sro = (long)BB * NR;          // 4096
  const long M_att = (long)BB * NO * NA;     // 3456
  const long M_obj = (long)BB * NO;          // 1152

  unsigned short* tsro_bf  = (unsigned short*)carve(M_sro * 3 * DD * 2);
  unsigned short* xattr_bf = (unsigned short*)carve(M_att * 2 * DD * 2);
  unsigned short* xobji_bf = (unsigned short*)carve(M_att * 2 * DD * 2);
  float* t_i = (float*)carve(M_att * DD * 4);
  float* t_e = (float*)carve(M_sro * DD * 4);
  unsigned short* ti_bf   = (unsigned short*)carve(M_att * DD * 2);
  unsigned short* te_bf   = (unsigned short*)carve(M_sro * DD * 2);
  unsigned short* obji_bf = (unsigned short*)carve(M_obj * DD * 2);
  unsigned short* obje_bf = (unsigned short*)carve(M_obj * DD * 2);
  float* p_i = (float*)carve(M_att * AAx * 4);
  float* p_e = (float*)carve(M_sro * AAx * 4);
  float* q_i = (float*)carve((long)HH * M_obj * AAx * 4);
  float* q_e = (float*)carve((long)HH * M_obj * AAx * 4);
  float* w_i = (float*)carve((long)BB * HH * NO * NA * 4);
  float* w_e = (float*)carve((long)BB * HH * NO * NR * 4);
  // packed bf16 weights
  unsigned short* Wp_rela = (unsigned short*)carve((long)3 * DD * DD * 2);
  unsigned short* Wp_obje = (unsigned short*)carve((long)3 * DD * DD * 2);
  unsigned short* Wp_attr = (unsigned short*)carve((long)2 * DD * DD * 2);
  unsigned short* Wp_obji = (unsigned short*)carve((long)2 * DD * DD * 2);
  unsigned short* Wp_ctxa = (unsigned short*)carve((long)DD * AAx * 2);
  unsigned short* Wp_ctxr = (unsigned short*)carve((long)DD * AAx * 2);
  unsigned short* Wp_ah   = (unsigned short*)carve((long)HH * DD * AAx * 2);
  unsigned short* Wp_rh   = (unsigned short*)carve((long)HH * DD * AAx * 2);

  // ---- pack activations ----
  {
    long tot = M_sro * 3 * DD;
    pack_tsro_kernel<<<cdiv(tot, 256), 256, 0, stream>>>(obj_vecs_e, rela_vecs, edges,
                                                         tsro_bf, (int)tot);
  }
  {
    long tot = M_att * 2 * DD;
    pack_attr_kernel<<<cdiv(tot, 256), 256, 0, stream>>>(attr_vecs, obj_vecs_i,
                                                         xattr_bf, xobji_bf, (int)tot);
  }
  // ---- pack weights into [K/16][N][16] bf16 ----
  auto packw = [&](const float* W, unsigned short* Wp, int K, int N, int Z) {
    long tot = (long)Z * K * N;
    pack_w_kernel<<<cdiv(tot, 256), 256, 0, stream>>>(W, Wp, K, N, tot);
  };
  packw(W_rela, Wp_rela, 3 * DD, DD, 1);
  packw(W_obje, Wp_obje, 3 * DD, DD, 1);
  packw(W_attr, Wp_attr, 2 * DD, DD, 1);
  packw(W_obji, Wp_obji, 2 * DD, DD, 1);
  packw(W_ctx_attr, Wp_ctxa, DD, AAx, 1);
  packw(W_ctx_rela, Wp_ctxr, DD, AAx, 1);
  packw(W_ah, Wp_ah, DD, AAx, HH);
  packw(W_rh, Wp_rh, DD, AAx, HH);
  // ---- bf16 copies of fp32 A-matrices ----
  cvt_bf16_kernel<<<cdiv(M_obj * DD, 256), 256, 0, stream>>>(obj_vecs_i, obji_bf, M_obj * DD);
  cvt_bf16_kernel<<<cdiv(M_obj * DD, 256), 256, 0, stream>>>(obj_vecs_e, obje_bf, M_obj * DD);

  // GEMM launchers (wave tile = 16x32, 8 waves/block)
  auto launch_gemm = [&](auto kern, const unsigned short* A, const unsigned short* Bp,
                         const float* bias, const float* resid, const float* rscale,
                         float* C, int M, int N, int K, int Z,
                         long bz, long biasz, long cz) {
    int tn2 = N / 32, tiles = (M / 16) * tn2;
    kern<<<dim3(cdiv(tiles, 8), 1, Z), 256, 0, stream>>>(
        A, Bp, bias, resid, rscale, C, N, K, tn2, tiles, bz, biasz, cz);
  };

  // ---- big GEMMs ----
  // new_rela = (relu(t_sro @ W_rela + b) + rela_vecs) * rela_masks -> d_out
  launch_gemm(gemm_wmma_kernel<1, 1, 1>, tsro_bf, Wp_rela, b_rela, rela_vecs,
              rela_masks, out_rela, (int)M_sro, DD, 3 * DD, 1, 0, 0, 0);
  // t_e = relu(t_sro @ W_obje + b)
  launch_gemm(gemm_wmma_kernel<1, 0, 0>, tsro_bf, Wp_obje, b_obje, nullptr, nullptr,
              t_e, (int)M_sro, DD, 3 * DD, 1, 0, 0, 0);
  // new_attr = relu(x_attr @ W_attr + b) + attr_vecs -> d_out
  launch_gemm(gemm_wmma_kernel<1, 1, 0>, xattr_bf, Wp_attr, b_attr, attr_vecs, nullptr,
              out_attr, (int)M_att, DD, 2 * DD, 1, 0, 0, 0);
  // t_i = relu(x_obji @ W_obji + b)
  launch_gemm(gemm_wmma_kernel<1, 0, 0>, xobji_bf, Wp_obji, b_obji, nullptr, nullptr,
              t_i, (int)M_att, DD, 2 * DD, 1, 0, 0, 0);

  // bf16 copies of t_i / t_e for the projection GEMMs
  cvt_bf16_kernel<<<cdiv(M_att * DD, 256), 256, 0, stream>>>(t_i, ti_bf, M_att * DD);
  cvt_bf16_kernel<<<cdiv(M_sro * DD, 256), 256, 0, stream>>>(t_e, te_bf, M_sro * DD);

  // ---- projection GEMMs ----
  launch_gemm(gemm_wmma_kernel<0, 0, 0>, ti_bf, Wp_ctxa, b_ctx_attr, nullptr, nullptr,
              p_i, (int)M_att, AAx, DD, 1, 0, 0, 0);
  launch_gemm(gemm_wmma_kernel<0, 0, 0>, te_bf, Wp_ctxr, b_ctx_rela, nullptr, nullptr,
              p_e, (int)M_sro, AAx, DD, 1, 0, 0, 0);
  launch_gemm(gemm_wmma_kernel<0, 0, 0>, obji_bf, Wp_ah, b_ah, nullptr, nullptr,
              q_i, (int)M_obj, AAx, DD, HH, (long)DD * AAx, (long)AAx, M_obj * (long)AAx);
  launch_gemm(gemm_wmma_kernel<0, 0, 0>, obje_bf, Wp_rh, b_rh, nullptr, nullptr,
              q_e, (int)M_obj, AAx, DD, HH, (long)DD * AAx, (long)AAx, M_obj * (long)AAx);

  // ---- attention logits (tanh + reduce over A) ----
  logit_i_kernel<<<BB * HH * NO, 128, 0, stream>>>(p_i, q_i, w_aa, b_aa, w_i);
  logit_e_kernel<<<BB * HH * NO, 256, 0, stream>>>(p_e, q_e, w_ra, b_ra, w_e);

  // ---- masked softmax + renorm ----
  softmax_attr_kernel<<<cdiv((long)BB * HH * NO, 256), 256, 0, stream>>>(attr_masks, w_i);
  softmax_rela_kernel<<<cdiv((long)BB * HH * NO * 32, 256), 256, 0, stream>>>(obj_rela_masks, w_e);

  // ---- weighted sums + residuals ----
  newobj_i_kernel<<<cdiv((long)BB * NO * DD, 256), 256, 0, stream>>>(w_i, t_i, obj_vecs_i, out_obji);
  newobj_e_kernel<<<BB * NO, 256, 0, stream>>>(w_e, t_e, obj_vecs_e, out_obje);
}
